// MSA_86955907875524
// MI455X (gfx1250) — matrix operand order repaired and checked by reading, not compiled, not run on hardware
//
#include <hip/hip_runtime.h>
#include <hip/hip_bf16.h>

// ---------------------------------------------------------------------------
// MI455X (gfx1250) attention pipeline, all matmuls on v_wmma_f32_16x16x32_bf16.
// B=8, S=1024, Dm=512, H=8, D_HEAD=64.
// ---------------------------------------------------------------------------

typedef __attribute__((ext_vector_type(16))) __bf16 v16bf;
typedef __attribute__((ext_vector_type(8)))  float  v8f;
typedef __attribute__((ext_vector_type(4)))  int    v4i;

union FragAB {            // 16 bf16 = 32 bytes = two float4 chunks
    v16bf  v;
    float4 q[2];
};

__device__ inline v8f v8f_zero() {
    v8f v = {0.f, 0.f, 0.f, 0.f, 0.f, 0.f, 0.f, 0.f};
    return v;
}

// ---- gfx1250 async global->LDS copy (ASYNCcnt path), with fallback ---------
#if defined(__has_builtin)
#if __has_builtin(__builtin_amdgcn_global_load_async_to_lds_b128)
#define HAVE_ASYNC_LDS 1
#endif
#endif

#ifdef HAVE_ASYNC_LDS
__device__ inline void async_copy16(const void* g, void* l) {
    // Signature (per clang diagnostic): (v4i AS1*, v4i AS3*, imm offset, imm cpol)
    __builtin_amdgcn_global_load_async_to_lds_b128(
        (__attribute__((address_space(1))) v4i*)g,
        (__attribute__((address_space(3))) v4i*)l, 0, 0);
}
__device__ inline void async_wait0() {
#if __has_builtin(__builtin_amdgcn_s_wait_asynccnt)
    __builtin_amdgcn_s_wait_asynccnt(0);
#else
    asm volatile("s_wait_asynccnt 0x0" ::: "memory");
#endif
}
#endif

// ---------------------------------------------------------------------------
// GEMM: Y[rows x 512] = X[rows x 512] @ W[512 x 512], bf16 WMMA, f32 accum.
// Block: 256 threads (8 waves). Block tile 128(M) x 64(N); wave tile 32x32.
// Staging uses b128 global loads + packed fp32->bf16 conversion.
// LDS rows padded to 40 bf16 (80 B): 16B-aligned frag reads, banks spread.
// ---------------------------------------------------------------------------
template <typename InT, typename OutT>
__global__ __launch_bounds__(256)
void gemm512(const InT* __restrict__ X, const float* __restrict__ W,
             OutT* __restrict__ Y) {
    constexpr int K = 512, N = 512;
    __shared__ alignas(16) __bf16 Abuf[128][40];   // [m][k]
    __shared__ alignas(16) __bf16 Bbuf[64][40];    // [n][k]  (transposed W tile)

    const int tid  = threadIdx.x;
    const int lane = tid & 31;
    const int w    = tid >> 5;
    const int half = lane >> 4;
    const int l16  = lane & 15;
    const int m0   = blockIdx.x * 128;
    const int n0   = blockIdx.y * 64;
    const int wm   = (w >> 1) * 32;
    const int wn   = (w & 1) * 32;

    v8f acc[2][2];
#pragma unroll
    for (int mi = 0; mi < 2; ++mi)
#pragma unroll
        for (int ni = 0; ni < 2; ++ni) acc[mi][ni] = v8f_zero();

    for (int k0 = 0; k0 < K; k0 += 32) {
        // ---- Stage A tile 128x32 -> bf16, vectorized b128 loads ----
        if (sizeof(InT) == 4) {
            const float* Xf = (const float*)X;
#pragma unroll
            for (int it = 0; it < 4; ++it) {
                int idx = it * 256 + tid;        // 0..1023 float4 chunks
                int i = idx >> 3;                // row 0..127
                int j = (idx & 7) * 4;           // col 0..28
                float4 x = *(const float4*)&Xf[(size_t)(m0 + i) * K + k0 + j];
                __bf16 h4[4] = {(__bf16)x.x, (__bf16)x.y, (__bf16)x.z,
                                (__bf16)x.w};
                __builtin_memcpy(&Abuf[i][j], h4, 8);
            }
        } else {
            const __bf16* Xh = (const __bf16*)X;
#pragma unroll
            for (int it = 0; it < 2; ++it) {
                int idx = it * 256 + tid;        // 0..511 x 8-bf16 chunks
                int i = idx >> 2;                // row 0..127
                int j = (idx & 3) * 8;           // col 0..24
                *(float4*)&Abuf[i][j] =
                    *(const float4*)&Xh[(size_t)(m0 + i) * K + k0 + j];
            }
        }
        // ---- Stage W tile 32x64 transposed: b128 loads, bf16 stores ----
#pragma unroll
        for (int it = 0; it < 2; ++it) {
            int idx = it * 256 + tid;            // 0..511 float4 chunks
            int kk = idx >> 4;                   // 0..31
            int nn = (idx & 15) * 4;             // 0..60
            float4 wv = *(const float4*)&W[(size_t)(k0 + kk) * N + n0 + nn];
            Bbuf[nn + 0][kk] = (__bf16)wv.x;
            Bbuf[nn + 1][kk] = (__bf16)wv.y;
            Bbuf[nn + 2][kk] = (__bf16)wv.z;
            Bbuf[nn + 3][kk] = (__bf16)wv.w;
        }
        // Prefetch next k-tile while this one is consumed (global_prefetch_b8).
        if (k0 + 32 < K) {
            __builtin_prefetch(
                (const char*)&X[(size_t)(m0 + (tid >> 1)) * K + k0 + 32] +
                    (tid & 1) * 64, 0, 3);
            __builtin_prefetch(&W[(size_t)(k0 + 32 + (tid >> 3)) * N + n0 +
                                  (tid & 7) * 8], 0, 3);
        }
        __syncthreads();

        FragAB a[2], bfr[2];
#pragma unroll
        for (int mi = 0; mi < 2; ++mi) {
            // A layout: row = lane%16, K = (e/8)*16 + half*8 + e%8
            const __bf16* ap = &Abuf[wm + mi * 16 + l16][half * 8];
            a[mi].q[0] = *(const float4*)ap;
            a[mi].q[1] = *(const float4*)(ap + 16);
        }
#pragma unroll
        for (int ni = 0; ni < 2; ++ni) {
            // B layout: N = lane%16, K = (e/8)*16 + half*8 + e%8
            const __bf16* bp = &Bbuf[wn + ni * 16 + l16][half * 8];
            bfr[ni].q[0] = *(const float4*)bp;
            bfr[ni].q[1] = *(const float4*)(bp + 16);
        }
#pragma unroll
        for (int mi = 0; mi < 2; ++mi)
#pragma unroll
            for (int ni = 0; ni < 2; ++ni)
                acc[mi][ni] = __builtin_amdgcn_wmma_f32_16x16x32_bf16(
                    false, a[mi].v, false, bfr[ni].v,
                    (short)0, acc[mi][ni], false, false);
        __syncthreads();
    }

    // Epilogue: C layout M = r + half*8, N = lane%16.
#pragma unroll
    for (int mi = 0; mi < 2; ++mi)
#pragma unroll
        for (int ni = 0; ni < 2; ++ni)
#pragma unroll
            for (int r = 0; r < 8; ++r) {
                int row = m0 + wm + mi * 16 + r + half * 8;
                int col = n0 + wn + ni * 16 + l16;
                Y[(size_t)row * N + col] = (OutT)acc[mi][ni][r];
            }
}

// ---------------------------------------------------------------------------
// Attention: grid (S/16, B), 256 threads = 8 waves; wave h handles head h for
// the block's 16 query rows. Flash-style online softmax over 32 keys/iter.
// scores_avg reduced across head-waves with ds_add_f32 into shared tile.
// V tiles staged via GLOBAL_LOAD_ASYNC_TO_LDS (ASYNCcnt) when available.
// ---------------------------------------------------------------------------
__global__ __launch_bounds__(256)
void attn_kernel(const __bf16* __restrict__ Qb, const __bf16* __restrict__ Kb,
                 const __bf16* __restrict__ Vb,
                 const unsigned char* __restrict__ mask,
                 const float* __restrict__ edge,
                 __bf16* __restrict__ Ctx, float* __restrict__ scores_avg) {
    constexpr int S = 1024, Dm = 512, HD = 64;
    __shared__ alignas(16) float  s_sc[16][32];       // head-summed scores tile
    __shared__ alignas(16) __bf16 s_p[8][16][40];     // per-wave P staging
#ifdef HAVE_ASYNC_LDS
    __shared__ alignas(16) __bf16 s_v[8][32][72];     // [key][d], 144B rows
#else
    __shared__ alignas(16) __bf16 s_vT[8][64][40];    // [d][key]
#endif

    const int tid  = threadIdx.x;
    const int lane = tid & 31;
    const int h    = tid >> 5;          // head = wave id
    const int half = lane >> 4;
    const int l16  = lane & 15;
    const int q0   = blockIdx.x * 16;
    const int b    = blockIdx.y;

    // Q A-fragments for head h (two K=32 slabs of d), direct 16B global loads.
    FragAB qa[2];
    {
        const __bf16* qrow = Qb + (size_t)(b * S + q0 + l16) * Dm + h * HD;
#pragma unroll
        for (int c = 0; c < 2; ++c) {
            const __bf16* p = qrow + c * 32 + half * 8;
            qa[c].q[0] = *(const float4*)p;
            qa[c].q[1] = *(const float4*)(p + 16);
        }
    }

    float m_r[8], l_r[8];
    v8f o[4];
#pragma unroll
    for (int n = 0; n < 4; ++n) o[n] = v8f_zero();
#pragma unroll
    for (int r = 0; r < 8; ++r) { m_r[r] = -3.0e38f; l_r[r] = 0.f; }

    for (int k0 = 0; k0 < S; k0 += 32) {
        // Zero the shared score-sum tile (16x32 f32 = 512 elems).
        ((float*)s_sc)[tid]       = 0.f;
        ((float*)s_sc)[tid + 256] = 0.f;
        __syncthreads();

        // Prefetch next tile's K/V rows (one cacheline per row segment).
        if (k0 + 32 < S) {
            const size_t nrow = (size_t)(b * S + k0 + 32 + lane) * Dm + h * HD;
            __builtin_prefetch(Kb + nrow, 0, 3);
            __builtin_prefetch(Vb + nrow, 0, 3);
        }

        // ---- Stage V tile (32 keys x 64 d) into this wave's slab ----
        {
            const __bf16* vbase = Vb + (size_t)(b * S + k0) * Dm + h * HD;
#ifdef HAVE_ASYNC_LDS
#pragma unroll
            for (int ch = 0; ch < 8; ++ch) {
                int idx = ch * 32 + lane;        // 0..255 -> 16B each
                int key = idx >> 3;              // 0..31
                int col = (idx & 7) * 8;         // 0..56
                async_copy16(vbase + (size_t)key * Dm + col, &s_v[h][key][col]);
            }
            async_wait0();
#else
#pragma unroll
            for (int ch = 0; ch < 8; ++ch) {
                int idx = ch * 32 + lane;
                int key = idx >> 3;
                int col = (idx & 7) * 8;
                float4 v4 = *(const float4*)(vbase + (size_t)key * Dm + col);
                const __bf16* vv = (const __bf16*)&v4;
#pragma unroll
                for (int i = 0; i < 8; ++i) s_vT[h][col + i][key] = vv[i];
            }
#endif
        }

        // S = Q K^T for 32 keys: 2 tiles x 2 d-slabs = 4 WMMAs.
        v8f sc[2];
#pragma unroll
        for (int t = 0; t < 2; ++t) {
            v8f a = v8f_zero();
#pragma unroll
            for (int c = 0; c < 2; ++c) {
                FragAB kb;   // B[d,key]: lane reads two 16B chunks of K row, 32 B apart
                const __bf16* kp = Kb +
                    (size_t)(b * S + k0 + t * 16 + l16) * Dm +
                    h * HD + c * 32 + half * 8;
                kb.q[0] = *(const float4*)kp;
                kb.q[1] = *(const float4*)(kp + 16);
                a = __builtin_amdgcn_wmma_f32_16x16x32_bf16(
                    false, qa[c].v, false, kb.v, (short)0, a, false, false);
            }
            sc[t] = a;
        }

        // Scale + edge bias + mask; accumulate head-sum for scores_avg.
#pragma unroll
        for (int t = 0; t < 2; ++t)
#pragma unroll
            for (int r = 0; r < 8; ++r) {
                int qrow = q0 + r + half * 8;
                int kcol = k0 + t * 16 + l16;
                size_t eidx = ((size_t)b * S + qrow) * S + kcol;
                float sv = mask[eidx] ? -1.0e9f
                                      : sc[t][r] * 0.125f + edge[eidx];
                sc[t][r] = sv;
                atomicAdd(&s_sc[r + half * 8][t * 16 + l16], sv);  // ds_add_f32
            }
        __syncthreads();

        // Write scores_avg tile (sum over heads / 8), coalesced.
#pragma unroll
        for (int it = 0; it < 2; ++it) {
            int idx = it * 256 + tid;
            int i = idx >> 5, j = idx & 31;
            scores_avg[((size_t)b * S + q0 + i) * S + k0 + j] =
                s_sc[i][j] * 0.125f;
        }
        __syncthreads();

        // Online softmax (rows live across 16 lanes of each half-wave).
#pragma unroll
        for (int r = 0; r < 8; ++r) {
            float mx = fmaxf(sc[0][r], sc[1][r]);
#pragma unroll
            for (int off = 8; off >= 1; off >>= 1)
                mx = fmaxf(mx, __shfl_xor(mx, off, 32));
            float mn = fmaxf(m_r[r], mx);
            float sf = __expf(m_r[r] - mn);
            float p0 = __expf(sc[0][r] - mn);
            float p1 = __expf(sc[1][r] - mn);
            float rs = p0 + p1;
#pragma unroll
            for (int off = 8; off >= 1; off >>= 1)
                rs += __shfl_xor(rs, off, 32);
            l_r[r] = l_r[r] * sf + rs;
            m_r[r] = mn;
#pragma unroll
            for (int n = 0; n < 4; ++n) o[n][r] *= sf;
            // Park P in wave-private LDS (C-layout -> A-layout transpose).
            s_p[h][r + half * 8][l16]      = (__bf16)p0;
            s_p[h][r + half * 8][16 + l16] = (__bf16)p1;
        }

        // O += P @ V : A = P (16x32), B slabs from the V tile -> 4 WMMAs.
        FragAB pa;
        {
            const __bf16* pp = &s_p[h][l16][half * 16];
            pa.q[0] = *(const float4*)pp;
            pa.q[1] = *(const float4*)(pp + 8);
        }
#pragma unroll
        for (int n = 0; n < 4; ++n) {
            FragAB vb;   // B[key, d]: element e -> key = half*16+e, d = n*16+l16
#ifdef HAVE_ASYNC_LDS
#pragma unroll
            for (int e = 0; e < 16; ++e)
                vb.v[e] = s_v[h][half * 16 + e][n * 16 + l16];
#else
            const __bf16* vp = &s_vT[h][n * 16 + l16][half * 16];
            vb.q[0] = *(const float4*)vp;
            vb.q[1] = *(const float4*)(vp + 8);
#endif
            o[n] = __builtin_amdgcn_wmma_f32_16x16x32_bf16(
                false, pa.v, false, vb.v, (short)0, o[n], false, false);
        }
        __syncthreads();   // protect s_sc zeroing of next iteration
    }

    // Normalize and store context (bf16) for the output projection.
#pragma unroll
    for (int n = 0; n < 4; ++n)
#pragma unroll
        for (int r = 0; r < 8; ++r) {
            int qrow = q0 + r + half * 8;
            Ctx[(size_t)(b * S + qrow) * Dm + h * HD + n * 16 + l16] =
                (__bf16)(o[n][r] / l_r[r]);
        }
}

// ---------------------------------------------------------------------------
// Launch: 3 projection GEMMs -> attention -> output GEMM.
// Workspace: Qb/Kb/Vb/Ctx bf16, 4 x 8192x512x2 B = 32 MB.
// ---------------------------------------------------------------------------
extern "C" void kernel_launch(void* const* d_in, const int* in_sizes, int n_in,
                              void* d_out, int out_size, void* d_ws,
                              size_t ws_size, hipStream_t stream) {
    constexpr int B = 8, S = 1024, Dm = 512;
    const float*         inQ  = (const float*)d_in[0];
    const float*         inK  = (const float*)d_in[1];
    const float*         inV  = (const float*)d_in[2];
    const unsigned char* msk  = (const unsigned char*)d_in[3];  // bool mask
    const float*         edge = (const float*)d_in[4];
    const float*         WQ   = (const float*)d_in[5];
    const float*         WK   = (const float*)d_in[6];
    const float*         WV   = (const float*)d_in[7];
    const float*         Wfc  = (const float*)d_in[8];

    float* out        = (float*)d_out;                       // [B,S,Dm]
    float* scores_avg = out + (size_t)B * S * Dm;            // [B,S,S]

    const size_t NROW = (size_t)B * S;                       // 8192
    __bf16* Qb = (__bf16*)d_ws;
    __bf16* Kb = Qb + NROW * Dm;
    __bf16* Vb = Kb + NROW * Dm;
    __bf16* Cb = Vb + NROW * Dm;

    dim3 gg(64, 8);   // 8192/128 M-tiles x 512/64 N-tiles
    dim3 bb(256);
    gemm512<float, __bf16><<<gg, bb, 0, stream>>>(inQ, WQ, Qb);
    gemm512<float, __bf16><<<gg, bb, 0, stream>>>(inK, WK, Kb);
    gemm512<float, __bf16><<<gg, bb, 0, stream>>>(inV, WV, Vb);

    attn_kernel<<<dim3(S / 16, B), 256, 0, stream>>>(Qb, Kb, Vb, msk, edge,
                                                     Cb, scores_avg);

    gemm512<__bf16, float><<<gg, bb, 0, stream>>>(Cb, Wfc, out);
}
